// CEEnhancedLSTM_2731599200917
// MI455X (gfx1250) — matrix operand order repaired and checked by reading, not compiled, not run on hardware
//
#include <hip/hip_runtime.h>

// ---------------------------------------------------------------------------
// CE-enhanced LSTM for MI455X (gfx1250, wave32, WMMA).
//  * Weights converted fp32->f16 once (L2-resident, ~25MB of 192MB L2).
//  * x@Wx fused with h@Wh as one [x_t | h] @ W_gates GEMM per step.
//  * All GEMMs: v_wmma_f32_16x16x32_f16, f32 accumulate, fused epilogues.
//  * A-tile staged with CDNA5 async copy (global_load_async_to_lds_b128,
//    ASYNCcnt); B-tile transposed into WMMA fragment layout via LDS scatter.
//  * 512 sequential steps chained on `stream` (graph replay amortizes).
// ---------------------------------------------------------------------------

typedef __attribute__((ext_vector_type(16))) _Float16 v16h;
typedef __attribute__((ext_vector_type(8)))  _Float16 v8h;
typedef __attribute__((ext_vector_type(8)))  float    v8f;

constexpr int kB = 128, kS = 512, kIn = 1024, kHid = 1024;
constexpr int kG = 4096;          // 4*HID
constexpr int kK = 2048;          // IN + HID
constexpr int kC1 = 512;          // HID/2
constexpr float kChi = 0.638f, kKappa = 0.35f, kDt = 0.01f;
constexpr int kNC = 6;
constexpr int LSTRIDE = 48;       // LDS row stride in halves (96B: 16B aligned frags)

struct EpiArgs {
  const float* bias;     // per-column bias
  float*       f0;       // gates / u1 / outputs base
  const float* c1f;      // evo: pre-evolution c (f32)
  const float* kap;      // evo: per-batch curvature scalar
  const float* ovec;     // evo: output gate
  _Float16*    h16;      // evo: m16 ; mirror: xh16 + kIn (stride kK)
  float*       cdst;     // evo: running c
  int          t;        // mirror: time index
};

// CDNA5 async memory->LDS copy, tracked by ASYNCcnt (no builtin; inline asm).
__device__ __forceinline__ void async_g2lds_b128(unsigned lds_off, const void* gaddr) {
  asm volatile("global_load_async_to_lds_b128 %0, %1, off"
               :: "v"(lds_off), "v"(gaddr) : "memory");
}
__device__ __forceinline__ void wait_asynccnt0() {
  asm volatile("s_wait_asynccnt 0" ::: "memory");
}

// One block: 256 threads (8 waves), tile M=128 x N=64; wave tile 32x32.
template <int EPI>
__global__ __launch_bounds__(256) void gemm_rnn(
    const _Float16* __restrict__ A, int lda,     // 128 x K, row-major f16
    const _Float16* __restrict__ Bm, int ldb,    // K x N, row-major f16
    int K, int N, EpiArgs ea)
{
  __shared__ _Float16 Al[128 * LSTRIDE];
  __shared__ _Float16 Bl[64 * LSTRIDE];

  const int tid  = threadIdx.x;
  const int lane = tid & 31;
  const int wave = tid >> 5;
  const int wr   = wave & 3;        // wave row: rows 32*wr .. +31
  const int wc   = wave >> 2;       // wave col: cols 32*wc .. +31
  const int half = lane >> 4;
  const int l15  = lane & 15;
  const int ka_lo = half * 8;       // A-frag K chunks {0..7,16..23} / {8..15,24..31}
  const int ka_hi = 16 + half * 8;
  const int kb_lo = half * 8;       // B-frag K chunks {0..7,16..23} / {8..15,24..31}
  const int nb   = blockIdx.x * 64;

  v8f acc[2][2] = {};

  for (int k0 = 0; k0 < K; k0 += 32) {
    __syncthreads();                 // protect LDS WAR from previous iteration
    // ---- stage A tile: 128 x 32 halves, async global->LDS (ASYNCcnt) ----
    {
      int r  = tid >> 1;
      int cc = (tid & 1) << 4;
      const _Float16* s = A + (size_t)r * lda + k0 + cc;
      async_g2lds_b128((unsigned)(uintptr_t)&Al[r * LSTRIDE + cc], s);
      async_g2lds_b128((unsigned)(uintptr_t)&Al[r * LSTRIDE + cc + 8], s + 8);
    }
    // ---- stage B tile transposed into fragment layout: Bl[n][k] ----
    {
      int fk = (tid * 8) >> 6;      // 0..31
      int fn = (tid * 8) & 63;      // 8 consecutive n
      const _Float16* s = Bm + (size_t)(k0 + fk) * ldb + nb + fn;
      v8h u = *(const v8h*)s;
#pragma unroll
      for (int j = 0; j < 8; ++j) Bl[(fn + j) * LSTRIDE + fk] = u[j];
    }
    if (k0 + 32 < K) {
      __builtin_prefetch(Bm + (size_t)(k0 + 32 + ((tid * 8) >> 6)) * ldb + nb, 0, 3);
    }
    wait_asynccnt0();                // A-tile async copies visible in LDS
    __syncthreads();

    // ---- build fragments from LDS, 4 WMMAs per K-step ----
    v16h af[2], bf[2];
#pragma unroll
    for (int mi = 0; mi < 2; ++mi) {
      int row = wr * 32 + mi * 16 + l15;
      v8h lo = *(const v8h*)&Al[row * LSTRIDE + ka_lo];
      v8h hi = *(const v8h*)&Al[row * LSTRIDE + ka_hi];
      af[mi] = __builtin_shufflevector(lo, hi, 0, 1, 2, 3, 4, 5, 6, 7,
                                       8, 9, 10, 11, 12, 13, 14, 15);
    }
#pragma unroll
    for (int ni = 0; ni < 2; ++ni) {
      int col = wc * 32 + ni * 16 + l15;
      v8h lo = *(const v8h*)&Bl[col * LSTRIDE + kb_lo];
      v8h hi = *(const v8h*)&Bl[col * LSTRIDE + kb_lo + 16];
      bf[ni] = __builtin_shufflevector(lo, hi, 0, 1, 2, 3, 4, 5, 6, 7,
                                       8, 9, 10, 11, 12, 13, 14, 15);
    }
#pragma unroll
    for (int mi = 0; mi < 2; ++mi)
#pragma unroll
      for (int ni = 0; ni < 2; ++ni)
        acc[mi][ni] = __builtin_amdgcn_wmma_f32_16x16x32_f16(
            false, af[mi], false, bf[ni], (short)0, acc[mi][ni], false, false);
  }

  // ---- fused epilogue (C layout: m = 8*half + v, n = l15 within 16x16) ----
#pragma unroll
  for (int mi = 0; mi < 2; ++mi) {
#pragma unroll
    for (int ni = 0; ni < 2; ++ni) {
      int n = nb + wc * 32 + ni * 16 + l15;
#pragma unroll
      for (int v = 0; v < 8; ++v) {
        int m = wr * 32 + mi * 16 + half * 8 + v;
        float r = acc[mi][ni][v] + ea.bias[n];
        if (EPI == 0) {                                   // gates: bias only
          ea.f0[(size_t)m * N + n] = r;
        } else if (EPI == 1) {                            // curvature: relu
          ea.f0[(size_t)m * N + n] = r > 0.f ? r : 0.f;
        } else if (EPI == 2) {                            // evolution of c
          size_t idx = (size_t)m * kHid + n;
          float c2 = ea.c1f[idx] + kDt * ea.kap[m] * r;
          ea.cdst[idx] = c2;
          ea.h16[idx] = (_Float16)(ea.ovec[idx] * tanhf(c2));
        } else {                                          // mirror -> h
          float h = tanhf(r);
          h = h / (1.f + kKappa * fabsf(h));
          ea.f0[((size_t)m * kS + ea.t) * kHid + n] = h;  // outputs[b,t,:]
          ea.h16[(size_t)m * kK + n] = (_Float16)h;       // next-step A operand
        }
      }
    }
  }
}

__global__ void cvt_f32_to_f16(const float* __restrict__ s, _Float16* __restrict__ d, int n) {
  int i = blockIdx.x * blockDim.x + threadIdx.x;
  if (i < n) d[i] = (_Float16)s[i];
}

__global__ void cvt_xt(const float* __restrict__ x, _Float16* __restrict__ xh, int t) {
  int i = blockIdx.x * blockDim.x + threadIdx.x;   // 0..B*IN-1
  int m = i >> 10, c = i & 1023;
  xh[(size_t)m * kK + c] = (_Float16)x[((size_t)m * kS + t) * kIn + c];
}

__global__ void zero_f16(_Float16* p, int n) {
  int i = blockIdx.x * blockDim.x + threadIdx.x;
  if (i < n) p[i] = (_Float16)0.f;
}
__global__ void zero_f32(float* p, int n) {
  int i = blockIdx.x * blockDim.x + threadIdx.x;
  if (i < n) p[i] = 0.f;
}

__global__ void lstm_cell(const float* __restrict__ gates, const float* __restrict__ cprev,
                          float* __restrict__ c1f, _Float16* __restrict__ c1h,
                          float* __restrict__ ovec) {
  int i = blockIdx.x * blockDim.x + threadIdx.x;   // 0..B*HID-1
  int m = i >> 10, n = i & 1023;
  const float* g = gates + (size_t)m * kG;
  float iv = 1.f / (1.f + expf(-g[n]));
  float fv = 1.f / (1.f + expf(-g[kHid + n]));
  float gv = tanhf(g[2 * kHid + n]);
  float ov = 1.f / (1.f + expf(-g[3 * kHid + n]));
  float c1 = fv * cprev[i] + iv * gv;
  c1f[i] = c1;
  c1h[i] = (_Float16)c1;
  ovec[i] = ov;
}

__global__ __launch_bounds__(256) void kap_reduce(
    const float* __restrict__ u1, const float* __restrict__ w2,
    const float* __restrict__ b2, float* __restrict__ kap) {
  __shared__ float red[256];
  int m = blockIdx.x, t = threadIdx.x;
  const float* u = u1 + (size_t)m * kC1;
  red[t] = u[t] * w2[t] + u[t + 256] * w2[t + 256];
  __syncthreads();
  for (int s = 128; s > 0; s >>= 1) {
    if (t < s) red[t] += red[t + s];
    __syncthreads();
  }
  if (t == 0) kap[m] = kChi * (red[0] + b2[0]);
}

__global__ void tail_copy(float* __restrict__ out, const float* __restrict__ cbuf) {
  int i = blockIdx.x * blockDim.x + threadIdx.x;   // 0..B*HID-1
  int m = i >> 10, n = i & 1023;
  size_t hoff = (size_t)kB * kS * kHid;
  out[hoff + i] = out[((size_t)m * kS + (kS - 1)) * kHid + n];  // final h
  out[hoff + (size_t)kB * kHid + i] = cbuf[i];                  // final c
}

__global__ void zeta_kernel(const float* __restrict__ zw, const float* __restrict__ zp,
                            float* __restrict__ out) {
  float dre = 0.f, dim = 0.f;
  for (int j = 0; j < kNC; ++j) {
    float L = 0.2f + 0.1f * (float)j;
    float w = 1.f / (1.f + expf(-zw[j]));
    float p = 1.f / (1.f + expf(-zp[j]));
    float e = expf(-0.5f * L);
    float c = cosf(2.f * L), s = sinf(2.f * L);
    float E1r = e * c, E1i = e * s;    // exp(-s0*L),     s0 = 0.5-2i
    float E2r = e * c, E2i = -e * s;   // exp(-(1-s0)*L)
    dre += w * 0.5f * ((E1r + p * E2r) - (E2r + p * E1r));
    dim += w * 0.5f * ((E1i + p * E2i) - (E2i + p * E1i));
  }
  out[0] = dre * dre + dim * dim;
}

extern "C" void kernel_launch(void* const* d_in, const int* in_sizes, int n_in,
                              void* d_out, int out_size, void* d_ws, size_t ws_size,
                              hipStream_t stream) {
  (void)in_sizes; (void)n_in; (void)out_size; (void)ws_size;
  const float* x    = (const float*)d_in[0];
  const float* Wg   = (const float*)d_in[1];
  const float* bg   = (const float*)d_in[2];
  const float* mirW = (const float*)d_in[3];
  const float* mirb = (const float*)d_in[4];
  const float* cW1  = (const float*)d_in[5];
  const float* cb1  = (const float*)d_in[6];
  const float* cW2  = (const float*)d_in[7];
  const float* cb2  = (const float*)d_in[8];
  const float* evoW = (const float*)d_in[9];
  const float* evob = (const float*)d_in[10];
  const float* zw   = (const float*)d_in[11];
  const float* zp   = (const float*)d_in[12];
  float* out = (float*)d_out;
  char* ws = (char*)d_ws;

  size_t off = 0;
  auto alloc = [&](size_t bytes) -> char* {
    char* p = ws + off;
    off += (bytes + 255) & ~(size_t)255;
    return p;
  };
  _Float16* Wg16  = (_Float16*)alloc((size_t)kK * kG * 2);      // 16 MB
  _Float16* mir16 = (_Float16*)alloc((size_t)kHid * kHid * 2);  // 2 MB
  _Float16* cu116 = (_Float16*)alloc((size_t)kHid * kC1 * 2);   // 1 MB
  _Float16* evo16 = (_Float16*)alloc((size_t)kHid * kHid * 2);  // 2 MB
  _Float16* xh16  = (_Float16*)alloc((size_t)kB * kK * 2);      // [x_t | h] f16
  _Float16* c1h   = (_Float16*)alloc((size_t)kB * kHid * 2);
  _Float16* m16   = (_Float16*)alloc((size_t)kB * kHid * 2);
  float* gates = (float*)alloc((size_t)kB * kG * 4);
  float* cbuf  = (float*)alloc((size_t)kB * kHid * 4);          // running c
  float* c1f   = (float*)alloc((size_t)kB * kHid * 4);
  float* ovec  = (float*)alloc((size_t)kB * kHid * 4);
  float* u1    = (float*)alloc((size_t)kB * kC1 * 4);
  float* kap   = (float*)alloc((size_t)kB * 4);

  // one-time weight conversion (stays L2-resident across the scan)
  cvt_f32_to_f16<<<(kK * kG + 255) / 256, 256, 0, stream>>>(Wg, Wg16, kK * kG);
  cvt_f32_to_f16<<<(kHid * kHid + 255) / 256, 256, 0, stream>>>(mirW, mir16, kHid * kHid);
  cvt_f32_to_f16<<<(kHid * kC1 + 255) / 256, 256, 0, stream>>>(cW1, cu116, kHid * kC1);
  cvt_f32_to_f16<<<(kHid * kHid + 255) / 256, 256, 0, stream>>>(evoW, evo16, kHid * kHid);
  zero_f16<<<(kB * kK + 255) / 256, 256, 0, stream>>>(xh16, kB * kK);     // h0 = 0
  zero_f32<<<(kB * kHid + 255) / 256, 256, 0, stream>>>(cbuf, kB * kHid); // c0 = 0

  for (int t = 0; t < kS; ++t) {
    cvt_xt<<<(kB * kIn) / 256, 256, 0, stream>>>(x, xh16, t);

    EpiArgs eg{bg, gates, nullptr, nullptr, nullptr, nullptr, nullptr, 0};
    gemm_rnn<0><<<kG / 64, 256, 0, stream>>>(xh16, kK, Wg16, kG, kK, kG, eg);

    lstm_cell<<<(kB * kHid) / 256, 256, 0, stream>>>(gates, cbuf, c1f, c1h, ovec);

    EpiArgs ec{cb1, u1, nullptr, nullptr, nullptr, nullptr, nullptr, 0};
    gemm_rnn<1><<<kC1 / 64, 256, 0, stream>>>(c1h, kHid, cu116, kC1, kHid, kC1, ec);

    kap_reduce<<<kB, 256, 0, stream>>>(u1, cW2, cb2, kap);

    EpiArgs ee{evob, nullptr, c1f, kap, ovec, m16, cbuf, 0};
    gemm_rnn<2><<<kHid / 64, 256, 0, stream>>>(c1h, kHid, evo16, kHid, kHid, kHid, ee);

    EpiArgs em{mirb, out, nullptr, nullptr, nullptr, xh16 + kIn, nullptr, t};
    gemm_rnn<3><<<kHid / 64, 256, 0, stream>>>(m16, kHid, mir16, kHid, kHid, kHid, em);
  }

  tail_copy<<<(kB * kHid) / 256, 256, 0, stream>>>(out, cbuf);
  zeta_kernel<<<1, 1, 0, stream>>>(zw, zp,
      out + (size_t)kB * kS * kHid + 2 * (size_t)kB * kHid);
}
